// GNNDecoders_67645734912700
// MI455X (gfx1250) — compile-verified, matrix operand order checked by reading, not run on hardware
//
#include <hip/hip_runtime.h>

typedef __attribute__((ext_vector_type(16))) _Float16 v16h;
typedef __attribute__((ext_vector_type(8)))  float    v8f;

#define EMB      300
#define N_NODES  100000
#define N_EDGES  400000
#define N_MASK   15000

// ---------------------------------------------------------------------------
// WMMA 16x16x32 f16 fragment helpers (wave32).
// Fragment layout (16-bit A-matrix; B given as rows of W since C = A@W^T):
//   lane l : matrix row/col = l % 16,  kHalf = 8*(l/16)
//   VGPR j : K pair = (j&4 ? 16 : 0) + kHalf + 2*(j&3)
// => lane needs f32 elements [base .. base+7] and [base+16 .. base+23],
//    base = k0 + kHalf.  All 16B aligned (k0 % 32 == 0, kHalf in {0,8},
//    rows are 300/600 floats so row bases stay 16B aligned).
// ---------------------------------------------------------------------------

__device__ __forceinline__ float4 prelu4(float4 v, float pa) {
    float4 r;
    r.x = v.x > 0.0f ? v.x : pa * v.x;
    r.y = v.y > 0.0f ? v.y : pa * v.y;
    r.z = v.z > 0.0f ? v.z : pa * v.z;
    r.w = v.w > 0.0f ? v.w : pa * v.w;
    return r;
}

__device__ __forceinline__ v16h cvt_frag(float4 a0, float4 a1, float4 a2, float4 a3) {
    v16h f;
    f[0]  = (_Float16)a0.x; f[1]  = (_Float16)a0.y;
    f[2]  = (_Float16)a0.z; f[3]  = (_Float16)a0.w;
    f[4]  = (_Float16)a1.x; f[5]  = (_Float16)a1.y;
    f[6]  = (_Float16)a1.z; f[7]  = (_Float16)a1.w;
    f[8]  = (_Float16)a2.x; f[9]  = (_Float16)a2.y;
    f[10] = (_Float16)a2.z; f[11] = (_Float16)a2.w;
    f[12] = (_Float16)a3.x; f[13] = (_Float16)a3.y;
    f[14] = (_Float16)a3.z; f[15] = (_Float16)a3.w;
    return f;
}

// Unconditional full-K fragment: 4x global_load_b128 (one clause).
__device__ __forceinline__ v16h load_frag_full(const float* __restrict__ row,
                                               int k0, int kHalf,
                                               float pa, bool doPrelu)
{
    const float4* rp4 = reinterpret_cast<const float4*>(row + k0 + kHalf);
    float4 a0 = rp4[0];   // base + 0..3
    float4 a1 = rp4[1];   // base + 4..7
    float4 a2 = rp4[4];   // base + 16..19
    float4 a3 = rp4[5];   // base + 20..23
    if (doPrelu) {
        a0 = prelu4(a0, pa); a1 = prelu4(a1, pa);
        a2 = prelu4(a2, pa); a3 = prelu4(a3, pa);
    }
    return cvt_frag(a0, a1, a2, a3);
}

// Guarded tail fragment (only for the final partial K chunk).
__device__ __forceinline__ v16h load_frag_tail(const float* __restrict__ row,
                                               int K, int kBase, int kHalf,
                                               float pa, bool doPrelu)
{
    v16h f;
    const float2* rp2 = reinterpret_cast<const float2*>(row);
#pragma unroll
    for (int j = 0; j < 8; ++j) {
        int k = kBase + ((j & 4) ? 16 : 0) + kHalf + 2 * (j & 3);
        float x0 = 0.0f, x1 = 0.0f;
        if (k < K) {                       // K is even: k<K implies k+1<K
            float2 t = rp2[k >> 1];
            x0 = t.x; x1 = t.y;
        }
        if (doPrelu) {
            x0 = x0 > 0.0f ? x0 : pa * x0;
            x1 = x1 > 0.0f ? x1 : pa * x1;
        }
        f[2 * j]     = (_Float16)x0;
        f[2 * j + 1] = (_Float16)x1;
    }
    return f;
}

// ---------------------------------------------------------------------------
// C[M,N] = op(act(A[M,K]) @ B[N,K]^T + bias).
// One wave computes a 16x64 strip (NT=4 accumulators). All sub-tiles are
// computed unconditionally (B rows clamped to N-1; columns >= N are garbage
// but never stored) so the hot loop is branch-free straight-line code and
// EXEC stays all-1s for every WMMA. N, K, NGROUPS are compile-time constants.
// ---------------------------------------------------------------------------
template <bool PRELU, bool RELU_OUT, bool BIAS, int N, int K, int NGROUPS>
__global__ __launch_bounds__(256) void gemm_wmma_kernel(
    const float* __restrict__ A, const float* __restrict__ B,
    const float* __restrict__ bias, float* __restrict__ C,
    int M, const float* __restrict__ preluP)
{
    constexpr int NT = 4;
    const int wave = threadIdx.x >> 5;
    const int lane = threadIdx.x & 31;
    const int tile = blockIdx.x * 8 + wave;
    const int tilesM = M >> 4;
    const int tileM = tile / NGROUPS;          // compile-time divisor
    const int group = tile % NGROUPS;
    if (tileM >= tilesM) return;               // wave-uniform: EXEC stays full

    const float pa = PRELU ? preluP[0] : 0.0f;
    const int kHalf = (lane >> 4) << 3;        // 0 or 8
    const int rowA = (tileM << 4) + (lane & 15);
    const float* aRow = A + (long long)rowA * K;

    const float* bRow[NT];
#pragma unroll
    for (int t = 0; t < NT; ++t) {
        int rb = ((group * NT + t) << 4) + (lane & 15);
        rb = rb < N ? rb : (N - 1);            // clamp: cols >= N never stored
        bRow[t] = B + (long long)rb * K;
    }

    v8f acc[NT] = {};

    constexpr int Kmain = K & ~31;
#pragma unroll 1
    for (int k0 = 0; k0 < Kmain; k0 += 32) {
        // Issue all 20 b128 loads up front (one clause), then cvt + 4 WMMAs.
        v16h a  = load_frag_full(aRow,    k0, kHalf, pa, PRELU);
        v16h b0 = load_frag_full(bRow[0], k0, kHalf, 0.0f, false);
        v16h b1 = load_frag_full(bRow[1], k0, kHalf, 0.0f, false);
        v16h b2 = load_frag_full(bRow[2], k0, kHalf, 0.0f, false);
        v16h b3 = load_frag_full(bRow[3], k0, kHalf, 0.0f, false);
        acc[0] = __builtin_amdgcn_wmma_f32_16x16x32_f16(
                     false, a, false, b0, (short)0, acc[0], false, false);
        acc[1] = __builtin_amdgcn_wmma_f32_16x16x32_f16(
                     false, a, false, b1, (short)0, acc[1], false, false);
        acc[2] = __builtin_amdgcn_wmma_f32_16x16x32_f16(
                     false, a, false, b2, (short)0, acc[2], false, false);
        acc[3] = __builtin_amdgcn_wmma_f32_16x16x32_f16(
                     false, a, false, b3, (short)0, acc[3], false, false);
    }
    if (Kmain < K) {
        v16h a  = load_frag_tail(aRow,    K, Kmain, kHalf, pa, PRELU);
        v16h b0 = load_frag_tail(bRow[0], K, Kmain, kHalf, 0.0f, false);
        v16h b1 = load_frag_tail(bRow[1], K, Kmain, kHalf, 0.0f, false);
        v16h b2 = load_frag_tail(bRow[2], K, Kmain, kHalf, 0.0f, false);
        v16h b3 = load_frag_tail(bRow[3], K, Kmain, kHalf, 0.0f, false);
        acc[0] = __builtin_amdgcn_wmma_f32_16x16x32_f16(
                     false, a, false, b0, (short)0, acc[0], false, false);
        acc[1] = __builtin_amdgcn_wmma_f32_16x16x32_f16(
                     false, a, false, b1, (short)0, acc[1], false, false);
        acc[2] = __builtin_amdgcn_wmma_f32_16x16x32_f16(
                     false, a, false, b2, (short)0, acc[2], false, false);
        acc[3] = __builtin_amdgcn_wmma_f32_16x16x32_f16(
                     false, a, false, b3, (short)0, acc[3], false, false);
    }

    // C/D layout: VGPR r -> M = r + 8*(lane/16), N = lane%16.
    const int mBase = (tileM << 4) + ((lane >> 4) << 3);
#pragma unroll
    for (int t = 0; t < NT; ++t) {
        int n = ((group * NT + t) << 4) + (lane & 15);
        if (n < N) {
            float bv = BIAS ? bias[n] : 0.0f;
#pragma unroll
            for (int r = 0; r < 8; ++r) {
                float v = acc[t][r] + bv;
                if (RELU_OUT) v = fmaxf(v, 0.0f);
                C[(long long)(mBase + r) * N + n] = v;
            }
        }
    }
}

// Zero out masked node rows of h (duplicates harmless: all writes are 0).
__global__ void mask_zero_kernel(const int* __restrict__ idx,
                                 float* __restrict__ h)
{
    int t = blockIdx.x * blockDim.x + threadIdx.x;
    if (t >= N_MASK * 75) return;
    int i = t / 75, q = t % 75;
    float4 z = {0.f, 0.f, 0.f, 0.f};
    reinterpret_cast<float4*>(h + (long long)idx[i] * EMB)[q] = z;
}

// aggr[i] = h[i] + E1[4] + E2[0]   (self-loop contribution, non-atomic init)
__global__ void aggr_init_kernel(const float* __restrict__ h,
                                 const float* __restrict__ E1,
                                 const float* __restrict__ E2,
                                 float* __restrict__ aggr)
{
    long long t = (long long)blockIdx.x * blockDim.x + threadIdx.x;
    if (t >= (long long)N_NODES * 75) return;
    long long i = t / 75; int q = (int)(t % 75);
    float4 a = reinterpret_cast<const float4*>(h)[i * 75 + q];
    float4 b = reinterpret_cast<const float4*>(E1 + 4 * EMB)[q];
    float4 c = reinterpret_cast<const float4*>(E2)[q];
    float4 o = {a.x + b.x + c.x, a.y + b.y + c.y,
                a.z + b.z + c.z, a.w + b.w + c.w};
    reinterpret_cast<float4*>(aggr)[i * 75 + q] = o;
}

// aggr[dst] += h[src] + E1[a0] + E2[a1]   via f32 atomics (L2 atomic units)
__global__ void edge_scatter_kernel(const float* __restrict__ h,
                                    const int* __restrict__ ei,
                                    const int* __restrict__ ea,
                                    const float* __restrict__ E1,
                                    const float* __restrict__ E2,
                                    float* __restrict__ aggr)
{
    long long t = (long long)blockIdx.x * blockDim.x + threadIdx.x;
    if (t >= (long long)N_EDGES * 75) return;
    int e = (int)(t / 75), q = (int)(t % 75);
    int src = ei[e];
    int dst = ei[N_EDGES + e];
    int a0 = ea[2 * e];
    int a1 = ea[2 * e + 1];
    float4 hv = reinterpret_cast<const float4*>(h  + (long long)src * EMB)[q];
    float4 e1 = reinterpret_cast<const float4*>(E1 + (long long)a0  * EMB)[q];
    float4 e2 = reinterpret_cast<const float4*>(E2 + (long long)a1  * EMB)[q];
    float* d = aggr + (long long)dst * EMB + q * 4;
    atomicAdd(d + 0, hv.x + e1.x + e2.x);
    atomicAdd(d + 1, hv.y + e1.y + e2.y);
    atomicAdd(d + 2, hv.z + e1.z + e2.z);
    atomicAdd(d + 3, hv.w + e1.w + e2.w);
}

extern "C" void kernel_launch(void* const* d_in, const int* in_sizes, int n_in,
                              void* d_out, int out_size, void* d_ws, size_t ws_size,
                              hipStream_t stream)
{
    const float* x          = (const float*)d_in[0];
    const int*   edge_index = (const int*)  d_in[1];
    const int*   edge_attr  = (const int*)  d_in[2];
    const int*   mask_idx   = (const int*)  d_in[3];
    const float* prelu_a    = (const float*)d_in[4];
    const float* W_enc      = (const float*)d_in[5];
    const float* E1         = (const float*)d_in[6];
    const float* E2         = (const float*)d_in[7];
    const float* W1         = (const float*)d_in[8];
    const float* b1         = (const float*)d_in[9];
    const float* W2         = (const float*)d_in[10];
    const float* b2         = (const float*)d_in[11];
    float*       out        = (float*)d_out;

    char* ws = (char*)d_ws;
    float* h      = (float*)(ws);                   // 100000*300 f32 = 120 MB
    float* aggr   = (float*)(ws + 120000000LL);     // 100000*300 f32 = 120 MB
    float* hidden = (float*)(ws + 240000000LL);     // 100000*600 f32 = 240 MB

    const int tilesM = N_NODES / 16;                // 6250 exactly

    // GEMM1: h = PReLU(x) @ W_enc^T   [100000,300]x[300,300], 5 col groups
    {
        long long tiles = (long long)tilesM * 5;
        int blocks = (int)((tiles + 7) / 8);
        gemm_wmma_kernel<true, false, false, EMB, EMB, 5><<<blocks, 256, 0, stream>>>(
            x, W_enc, nullptr, h, N_NODES, prelu_a);
    }

    // Zero masked rows of h
    {
        long long n = (long long)N_MASK * 75;
        mask_zero_kernel<<<(int)((n + 255) / 256), 256, 0, stream>>>(mask_idx, h);
    }

    // aggr = h + selfloop edge embedding
    {
        long long n = (long long)N_NODES * 75;
        aggr_init_kernel<<<(int)((n + 255) / 256), 256, 0, stream>>>(h, E1, E2, aggr);
    }

    // Scatter-add messages over 400k edges
    {
        long long n = (long long)N_EDGES * 75;
        edge_scatter_kernel<<<(int)((n + 255) / 256), 256, 0, stream>>>(
            h, edge_index, edge_attr, E1, E2, aggr);
    }

    // GEMM2: hidden = relu(aggr @ W1^T + b1)   [100000,300]x[600,300], 10 groups
    {
        long long tiles = (long long)tilesM * 10;
        int blocks = (int)((tiles + 7) / 8);
        gemm_wmma_kernel<false, true, true, 600, EMB, 10><<<blocks, 256, 0, stream>>>(
            aggr, W1, b1, hidden, N_NODES, nullptr);
    }

    // GEMM3: out = hidden @ W2^T + b2   [100000,600]x[300,600], 5 groups
    {
        long long tiles = (long long)tilesM * 5;
        int blocks = (int)((tiles + 7) / 8);
        gemm_wmma_kernel<false, false, true, EMB, 600, 5><<<blocks, 256, 0, stream>>>(
            hidden, W2, b2, out, N_NODES, nullptr);
    }
}
